// GraphEncoder_27539330302398
// MI455X (gfx1250) — matrix operand order repaired and checked by reading, not compiled, not run on hardware
//
#include <hip/hip_runtime.h>
#include <hip/hip_bf16.h>

// ---------------------------------------------------------------------------
// MI455X (gfx1250) GNN encoder.
//
// Roofline: dominant cost is 3 sweeps of adj (12000^2 fp32 = 576 MB, > 192MB
// L2) for adj@X with C=256/256/128: 2*N^2*640 ~= 184 GFLOP over ~1.7 GB HBM
// traffic -> ~128 flop/byte. At 23.3 TB/s the floor is ~74 us, so the compute
// side must sustain ~3 PFLOPS: only v_wmma_f32_16x16x32_bf16 gets there.
// adj tiles are converted f32->bf16 on the fly (values are U(0,1)/N; f32
// accumulation); B operands live in bf16 TRANSPOSED (C x K) layout so WMMA
// fragments are contiguous and load as ds_load_b128 pairs.
//
// Data movement:
//  - A (adj) tiles: float4 global loads, packed uint2 LDS stores directly in
//    WMMA A-fragment order, double buffered.
//  - B tiles: Tensor Data Mover (tensor_load_to_lds) with a D# describing a
//    32 x C tile of the (C x K) bf16 operand, LDS padding 16B per 64B row to
//    match fragment stride; issued by wave 0 for the NEXT K-step while all
//    waves run WMMAs on the current one; s_wait_tensorcnt + one barrier/iter.
//  - Each wave computes 16x128 (8 WMMA tiles) per K-step.
// ---------------------------------------------------------------------------

typedef __attribute__((ext_vector_type(16))) __bf16        v16bf;
typedef __attribute__((ext_vector_type(8)))  float         v8f;
typedef __attribute__((ext_vector_type(4)))  unsigned int  v4u;
typedef __attribute__((ext_vector_type(8)))  int           v8i;
typedef __attribute__((ext_vector_type(4)))  int           v4i;

__device__ __forceinline__ unsigned short f2bf(float f) {
  unsigned u = __builtin_bit_cast(unsigned, f);
  u += 0x7FFFu + ((u >> 16) & 1u);   // round-to-nearest-even
  return (unsigned short)(u >> 16);
}

__device__ __forceinline__ v16bf load_frag(const unsigned short* p) {
  union { uint4 q[2]; v16bf v; } u;
  u.q[0] = *(const uint4*)(p);
  u.q[1] = *(const uint4*)(p + 8);
  return u.v;
}

// TDM: load a (32 x rows) bf16 tile (contiguous-k rows) from transposed B
// (C x K) into LDS, 64B rows padded to 80B (pad_interval=16DW, pad_amount=4DW).
__device__ __forceinline__ void tdm_load_b(const unsigned short* src, int K, int k0,
                                           unsigned short* dstLds, int rows) {
  unsigned long long ga = (unsigned long long)(const void*)(src + k0);
  unsigned lds = (unsigned)(unsigned long long)(void*)dstLds;
  v4u g0;
  g0[0] = 1u;                                               // count=1, user D#
  g0[1] = lds;                                              // lds_addr
  g0[2] = (unsigned)ga;                                     // global_addr[31:0]
  g0[3] = (unsigned)((ga >> 32) & 0x1FFFFFFu) | (2u << 30); // addr[56:32]|type=2
  v8i g1;
  g1[0] = (1 << 16) | (1 << 20) | (3 << 22) | (3 << 25);    // 2B elems, pad 4DW/16DW
  g1[1] = (K & 0xFFFF) << 16;                               // tensor_dim0[15:0]
  g1[2] = ((K >> 16) & 0xFFFF) | ((rows & 0xFFFF) << 16);   // dim0[31:16], dim1[15:0]
  g1[3] = (32 << 16);                                       // dim1[31:16]=0, tile_dim0=32
  g1[4] = (rows & 0xFFFF);                                  // tile_dim1=rows, tile_dim2=0
  g1[5] = K;                                                // tensor_dim0_stride[31:0]
  g1[6] = 0;                                                // stride hi, dim1_stride lo
  g1[7] = 0;
  v4i z4 = {0, 0, 0, 0};
#if __clang_major__ >= 23
  v8i z8 = {0, 0, 0, 0, 0, 0, 0, 0};
  __builtin_amdgcn_tensor_load_to_lds(g0, g1, z4, z4, z8, 0);
#else
  __builtin_amdgcn_tensor_load_to_lds(g0, g1, z4, z4, 0);
#endif
}

// f32 (K x C) row-major -> bf16 transposed (C x K)
__global__ __launch_bounds__(256) void cvt_w_kernel(const float* __restrict__ in,
                                                    unsigned short* __restrict__ out,
                                                    int K, int C) {
  int idx = blockIdx.x * 256 + threadIdx.x;
  if (idx < K * C) {
    int r = idx / C, c = idx % C;
    out[(long)c * K + r] = f2bf(in[idx]);
  }
}

// out = act( A(f32, M x K) @ B(bf16, transposed C x K) + bias ).
// 256 threads = 8 waves arranged WM x WN (WM*WN==8); wave tile 16 x 128
// (8 wmma 16x16 tiles), K-step 32, double buffered. K % 32 == 0, C == 128*WN.
// OUT_BF16: write transposed bf16 (C x M); else row-major f32 (M x C).
template <int WM, int WN, bool OUT_BF16, bool RELU, bool BIAS>
__global__ __launch_bounds__(256) void gemm_bf16_kernel(
    const float* __restrict__ A, const unsigned short* __restrict__ B,
    const float* __restrict__ bias, void* __restrict__ outp,
    int M, int K, int C) {
  constexpr int BM  = 16 * WM;    // rows per block
  constexpr int BN  = 128 * WN;   // cols per block (== C)
  constexpr int NT  = 8;          // wmma tiles per wave (16x128)
  constexpr int STR = 40;         // LDS row stride in ushorts (64B data + 16B pad)

  __shared__ __align__(16) unsigned short As[2 * BM * STR];  // fragment order
  __shared__ __align__(16) unsigned short Bs[2 * BN * STR];  // [n][k 0..31]

  const int tid  = threadIdx.x;
  const int lane = tid & 31;
  const int wave = tid >> 5;
  const int wm   = wave / WN;
  const int wn   = wave % WN;
  const int half = lane >> 4;    // 0: lanes 0-15, 1: lanes 16-31
  const int l16  = lane & 15;
  const int m0   = blockIdx.x * BM;

  // Stage A tile (BM x 32) f32->bf16 into WMMA A-fragment order:
  // row i holds [half0: K 0..15][half1: K 16..31]; every aligned
  // group of 4 source K's lands contiguously -> float4 load + uint2 LDS store.
  auto stage_a = [&](int k0, unsigned short* dst) {
    for (int e = tid; e < BM * 8; e += 256) {
      int i = e >> 3, c = e & 7;
      int gm = m0 + i;
      float4 f = make_float4(0.f, 0.f, 0.f, 0.f);
      if (gm < M) f = *(const float4*)(A + (long)gm * K + k0 + c * 4);
      uint2 p;
      p.x = (unsigned)f2bf(f.x) | ((unsigned)f2bf(f.y) << 16);
      p.y = (unsigned)f2bf(f.z) | ((unsigned)f2bf(f.w) << 16);
      int pos = c << 2;
      *(uint2*)(dst + i * STR + pos) = p;
    }
  };

  v8f acc[NT];
#pragma unroll
  for (int t = 0; t < NT; ++t)
#pragma unroll
    for (int r = 0; r < 8; ++r) acc[t][r] = 0.0f;

  // Prologue: fill buffer 0.
  if (wave == 0) tdm_load_b(B, K, 0, Bs, BN);
  stage_a(0, As);
  if (wave == 0) __builtin_amdgcn_s_wait_tensorcnt(0);
  __syncthreads();

  int cur = 0;
  for (int k0 = 0; k0 < K; k0 += 32) {
    const unsigned short* Ac = As + cur * BM * STR;
    const unsigned short* Bc = Bs + cur * BN * STR;
    int nxt = k0 + 32;
    if (nxt < K) {  // prefetch next K-step into the other buffer
      if (wave == 0) tdm_load_b(B, K, nxt, Bs + (cur ^ 1) * BN * STR, BN);
      stage_a(nxt, As + (cur ^ 1) * BM * STR);
    }

    v16bf a = load_frag(Ac + (wm * 16 + l16) * STR + half * 16);
#pragma unroll
    for (int t = 0; t < NT; ++t) {
      v16bf b = load_frag(Bc + (wn * 128 + t * 16 + l16) * STR + half * 16);
      acc[t] = __builtin_amdgcn_wmma_f32_16x16x32_bf16(
          false, a, false, b, (short)0, acc[t], false, false);
    }

    if (wave == 0) __builtin_amdgcn_s_wait_tensorcnt(0);
    __syncthreads();
    cur ^= 1;
  }

  // Epilogue: C/D layout: VGPR r, lanes0-15 -> M=r, lanes16-31 -> M=8+r.
#pragma unroll
  for (int t = 0; t < NT; ++t) {
    int col = wn * 128 + t * 16 + l16;
    float bv = BIAS ? bias[col] : 0.0f;
#pragma unroll
    for (int r = 0; r < 8; ++r) {
      int grow = m0 + wm * 16 + half * 8 + r;
      if (grow < M) {
        float v = acc[t][r] + bv;
        if (RELU) v = fmaxf(v, 0.0f);
        if (OUT_BF16)
          ((unsigned short*)outp)[(long)col * M + grow] = f2bf(v);
        else
          ((float*)outp)[(long)grow * C + col] = v;
      }
    }
  }
}

// out = relu( LN(raw + res) * g + be ), C fixed = 256, one block per row.
__global__ __launch_bounds__(256) void ln_relu_kernel(
    const float* __restrict__ raw, const float* __restrict__ res,
    const float* __restrict__ g, const float* __restrict__ be,
    float* __restrict__ out) {
  __shared__ float sh[256];
  const int row = blockIdx.x, tid = threadIdx.x;
  float h = raw[(long)row * 256 + tid] + res[(long)row * 256 + tid];
  sh[tid] = h;
  __syncthreads();
  for (int s = 128; s > 0; s >>= 1) {
    if (tid < s) sh[tid] += sh[tid + s];
    __syncthreads();
  }
  float mu = sh[0] * (1.0f / 256.0f);
  __syncthreads();
  float d = h - mu;
  sh[tid] = d * d;
  __syncthreads();
  for (int s = 128; s > 0; s >>= 1) {
    if (tid < s) sh[tid] += sh[tid + s];
    __syncthreads();
  }
  float var = sh[0] * (1.0f / 256.0f);
  float v = d * rsqrtf(var + 1e-5f) * g[tid] + be[tid];
  out[(long)row * 256 + tid] = fmaxf(v, 0.0f);
}

extern "C" void kernel_launch(void* const* d_in, const int* in_sizes, int n_in,
                              void* d_out, int out_size, void* d_ws, size_t ws_size,
                              hipStream_t stream) {
  (void)in_sizes; (void)n_in; (void)out_size; (void)ws_size;
  const float* x   = (const float*)d_in[0];
  const float* adj = (const float*)d_in[1];
  const float* W1  = (const float*)d_in[2];
  const float* b1  = (const float*)d_in[3];
  const float* g1  = (const float*)d_in[4];
  const float* be1 = (const float*)d_in[5];
  const float* W2  = (const float*)d_in[6];
  const float* b2  = (const float*)d_in[7];
  const float* g2  = (const float*)d_in[8];
  const float* be2 = (const float*)d_in[9];
  const float* W3  = (const float*)d_in[10];
  const float* b3  = (const float*)d_in[11];
  const float* Wp1 = (const float*)d_in[12];
  const float* bp1 = (const float*)d_in[13];
  const float* Wp2 = (const float*)d_in[14];
  const float* bp2 = (const float*)d_in[15];

  const int N = 12000, IN = 256, HID = 256, EMB = 128, PROJ = 128;

  size_t off = 0;
  char* wsb = (char*)d_ws;
  auto alloc = [&](size_t bytes) -> void* {
    void* p = wsb + off;
    off += (bytes + 255) & ~(size_t)255;
    return p;
  };
  unsigned short* W1b  = (unsigned short*)alloc((size_t)IN  * HID  * 2);
  unsigned short* W2b  = (unsigned short*)alloc((size_t)HID * HID  * 2);
  unsigned short* W3b  = (unsigned short*)alloc((size_t)HID * EMB  * 2);
  unsigned short* Wp1b = (unsigned short*)alloc((size_t)EMB * PROJ * 2);
  unsigned short* Wp2b = (unsigned short*)alloc((size_t)PROJ* PROJ * 2);
  unsigned short* T    = (unsigned short*)alloc((size_t)N * HID * 2); // transposed C x N bf16
  float* Hraw = (float*)alloc((size_t)N * HID * 4);
  float* H1   = (float*)alloc((size_t)N * HID * 4);
  float* H2   = (float*)alloc((size_t)N * HID * 4);
  float* Z1   = (float*)alloc((size_t)N * PROJ * 4);

  float* z_out   = (float*)d_out;
  float* emb_out = (float*)d_out + (size_t)N * PROJ;

  dim3 blk(256);
  // weights -> bf16 transposed
  cvt_w_kernel<<<(IN * HID + 255) / 256, blk, 0, stream>>>(W1, W1b, IN, HID);
  cvt_w_kernel<<<(HID * HID + 255) / 256, blk, 0, stream>>>(W2, W2b, HID, HID);
  cvt_w_kernel<<<(HID * EMB + 255) / 256, blk, 0, stream>>>(W3, W3b, HID, EMB);
  cvt_w_kernel<<<(EMB * PROJ + 255) / 256, blk, 0, stream>>>(Wp1, Wp1b, EMB, PROJ);
  cvt_w_kernel<<<(PROJ * PROJ + 255) / 256, blk, 0, stream>>>(Wp2, Wp2b, PROJ, PROJ);

  const int g64  = (N + 63) / 64;    // <4,2> tiles: 64 rows x 256 cols
  const int g128 = (N + 127) / 128;  // <8,1> tiles: 128 rows x 128 cols

  // layer 1: T1 = x@W1+b1 (bf16^T), H = adj@T1, h1 = relu(LN(H + x))
  gemm_bf16_kernel<4, 2, true,  false, true ><<<g64, blk, 0, stream>>>(x,   W1b, b1, T,    N, IN, HID);
  gemm_bf16_kernel<4, 2, false, false, false><<<g64, blk, 0, stream>>>(adj, T, nullptr, Hraw, N, N, HID);
  ln_relu_kernel<<<N, blk, 0, stream>>>(Hraw, x, g1, be1, H1);

  // layer 2
  gemm_bf16_kernel<4, 2, true,  false, true ><<<g64, blk, 0, stream>>>(H1,  W2b, b2, T,    N, HID, HID);
  gemm_bf16_kernel<4, 2, false, false, false><<<g64, blk, 0, stream>>>(adj, T, nullptr, Hraw, N, N, HID);
  ln_relu_kernel<<<N, blk, 0, stream>>>(Hraw, H1, g2, be2, H2);

  // embedding = adj @ (h2@W3+b3)  (f32, straight into d_out)
  gemm_bf16_kernel<8, 1, true,  false, true ><<<g128, blk, 0, stream>>>(H2,  W3b, b3, T,      N, HID, EMB);
  gemm_bf16_kernel<8, 1, false, false, false><<<g128, blk, 0, stream>>>(adj, T, nullptr, emb_out, N, N, EMB);

  // projection head: z = relu(emb@Wp1+bp1)@Wp2+bp2
  gemm_bf16_kernel<8, 1, false, true,  true ><<<g128, blk, 0, stream>>>(emb_out, Wp1b, bp1, Z1,    N, EMB, PROJ);
  gemm_bf16_kernel<8, 1, false, false, true ><<<g128, blk, 0, stream>>>(Z1,      Wp2b, bp2, z_out, N, PROJ, PROJ);
}
